// FluxGNN1DLatent_41807211659831
// MI455X (gfx1250) — matrix operand order repaired and checked by reading, not compiled, not run on hardware
//
#include <hip/hip_runtime.h>
#include <hip/hip_bf16.h>

// ---------------------------------------------------------------------------
// FluxGNN1DLatent on MI455X (gfx1250, wave32, WMMA)
//
// out[b,n] = u[b,n]*S - (DT/DX)*(fluxdot[b,n+1]-fluxdot[b,n])
//   S          = sum(enc^2)/(sum(enc^2)+1e-12)
//   fluxdot[i] = sum_k tanh(MLP(feat_i))_k * 0.25 * dec[k]
//   feat_i     = [(u_l+u_r)*enc , |u_r-u_l|*|enc|]  (64 dims)
//
// Pass 1: one wave per block, 16 interfaces per tile, 20x
//         v_wmma_f32_16x16x32_f16 per tile; LDS round-trip for the
//         D-layout -> A-layout transpose between MLP layers.
// Pass 2: elementwise finalize.
//
// Activation math is branch-free: erf via A&S 7.1.26 (|err|<=1.5e-7, far
// below f16 GEMM noise), tanh via the gfx1250 V_TANH_F32 TRANS op.
//
// NOTE: abs/copysign must be OPAQUE (inline asm v_and_b32 / v_bfi_b32).
// Any source form InstCombine can recognize as llvm.fabs gets fused as
// fabs(x)*fabs(y)->fabs(x*y), SLP-vectorized to v2f32, and crashes ISel on
// this toolchain (v2f32<->i32 bitcast around the integer-AND fabs idiom).
// ---------------------------------------------------------------------------

typedef _Float16 half_t;
typedef __attribute__((ext_vector_type(16))) _Float16 v16h;
typedef __attribute__((ext_vector_type(8)))  _Float16 v8h;
typedef __attribute__((ext_vector_type(8)))  float    v8f;

#define B_SZ 32
#define N_SZ 65536
#define NI   (N_SZ + 1)          // 65537 interfaces per batch row
#define LD   32
#define HID  64
#define TILES_PER_B ((NI + 15) / 16)   // 4097
#define N_TILES (B_SZ * TILES_PER_B)   // 131104

union V16 { v16h v; v8h h[2]; };

__device__ __forceinline__ float abs_opaque(float x) {
    float r;
    asm("v_and_b32 %0, 0x7fffffff, %1" : "=v"(r) : "v"(x));
    return r;
}

__device__ __forceinline__ float copysign_opaque(float mag, float sgn) {
    // D = (0x7fffffff & mag) | (~0x7fffffff & sgn): magnitude of mag, sign of sgn
    float r;
    asm("v_bfi_b32 %0, 0x7fffffff, %1, %2" : "=v"(r) : "v"(mag), "v"(sgn));
    return r;
}

__device__ __forceinline__ float tanh_hw(float x) {
    // gfx1250 V_TANH_F32 (CDNA5 transcendental)
    float r;
    asm("v_tanh_f32 %0, %1" : "=v"(r) : "v"(x));
    return r;
}

__device__ __forceinline__ float erf_fast(float x) {
    // Abramowitz & Stegun 7.1.26, |abs err| <= 1.5e-7, branch-free.
    float ax = abs_opaque(x);
    float t  = __builtin_amdgcn_rcpf(fmaf(0.3275911f, ax, 1.0f));
    float p  = fmaf(1.061405429f, t, -1.453152027f);
    p = fmaf(p, t, 1.421413741f);
    p = fmaf(p, t, -0.284496736f);
    p = fmaf(p, t, 0.254829592f);
    p = p * t;
    float e  = __builtin_amdgcn_exp2f(ax * ax * -1.4426950408889634f); // exp(-ax^2)
    float er = fmaf(-p, e, 1.0f);
    return copysign_opaque(er, x);
}

__device__ __forceinline__ float gelu_exact(float x) {
    return 0.5f * x * (1.0f + erf_fast(x * 0.70710678118654752440f));
}

__device__ __forceinline__ v8f wmma_f16(v16h a, v16h b, v8f c) {
    // (neg_a, A, neg_b, B, c_mod, C, reuse_a, reuse_b)
    return __builtin_amdgcn_wmma_f32_16x16x32_f16(false, a, false, b,
                                                  (short)0, c, false, false);
}

__global__ __launch_bounds__(32) void flux_mlp_pass1(
    const float* __restrict__ u,  const float* __restrict__ enc,
    const float* __restrict__ W1, const float* __restrict__ b1,
    const float* __restrict__ W2, const float* __restrict__ b2,
    const float* __restrict__ W3, const float* __restrict__ b3,
    float* __restrict__ fluxdot)
{
    __shared__ alignas(16) half_t xbuf[16 * HID];   // 2 KB staging for transpose

    const int lane = (int)(threadIdx.x & 31u);
    const int ln   = lane & 15;      // column/row-within-half index
    const int hh   = lane >> 4;      // lane half (0 or 1)

    // ---- Build weight B-tiles (f32 -> f16, B 32x16 VGPR layout) ----
    // B layout: lane gives N = n0+ln; element j (0..15) gives K = k0 + 8*hh + (j&7) + (j>=8 ? 16 : 0).
    v16h B1t[2][4], B2t[2][4], B3t[2][2];
    #pragma unroll
    for (int kt = 0; kt < 2; ++kt) {
        #pragma unroll
        for (int nt = 0; nt < 4; ++nt) {
            V16 t1, t2;
            #pragma unroll
            for (int j = 0; j < 16; ++j) {
                int k = kt * 32 + 8 * hh + (j & 7) + ((j >> 3) << 4);
                int n = nt * 16 + ln;
                t1.v[j] = (half_t)W1[k * HID + n];
                t2.v[j] = (half_t)W2[k * HID + n];
            }
            B1t[kt][nt] = t1.v;
            B2t[kt][nt] = t2.v;
        }
        #pragma unroll
        for (int nt = 0; nt < 2; ++nt) {
            V16 t;
            #pragma unroll
            for (int j = 0; j < 16; ++j) {
                int k = kt * 32 + 8 * hh + (j & 7) + ((j >> 3) << 4);
                t.v[j] = (half_t)W3[k * LD + nt * 16 + ln];
            }
            B3t[kt][nt] = t.v;
        }
    }

    // ---- enc pattern for building A tiles in-register ----
    // A 16x32 f16 layout: element j (0..15) -> K = 8*hh + (j&7) + (j>=8 ? 16 : 0)
    float ef[16], efa[16];
    #pragma unroll
    for (int j = 0; j < 16; ++j) {
        int k = 8 * hh + (j & 7) + ((j >> 3) << 4);
        float e = enc[k];
        ef[j]  = e;
        efa[j] = abs_opaque(e);
    }

    // Per-lane biases matched to the D 16x16 f32 layout (N = tile*16 + ln)
    float b1v[4], b2v[4], b3v[2];
    #pragma unroll
    for (int nt = 0; nt < 4; ++nt) {
        b1v[nt] = b1[nt * 16 + ln];
        b2v[nt] = b2[nt * 16 + ln];
    }
    b3v[0] = b3[ln];
    b3v[1] = b3[16 + ln];

    // dec = enc / (sum(enc^2)+1e-12); FLUX_SCALE=0.25 folded in at use
    float s2 = 0.f;
    #pragma unroll
    for (int k = 0; k < LD; ++k) { float e = enc[k]; s2 += e * e; }
    float inv = 1.0f / (s2 + 1e-12f);
    float decv0 = enc[ln] * inv;
    float decv1 = enc[16 + ln] * inv;

    for (int tile = blockIdx.x; tile < N_TILES; tile += gridDim.x) {
        int b  = tile / TILES_PER_B;
        int i0 = (tile - b * TILES_PER_B) * 16;
        int i  = i0 + ln;                       // this lane's interface (row M=ln)
        int iv = (i < NI - 1) ? i : (NI - 1);   // clamp OOB rows of last tile
        const float* ub = u + (size_t)b * N_SZ;
        float ul = ub[(iv > 0) ? (iv - 1) : 0];          // edge-padded left
        float ur = ub[(iv < N_SZ) ? iv : (N_SZ - 1)];    // edge-padded right
        float fa = ul + ur;
        float fd = abs_opaque(ur - ul);

        // A tiles for layer 1 built directly in registers
        V16 A0, A1;
        #pragma unroll
        for (int j = 0; j < 16; ++j) {
            A0.v[j] = (half_t)(fa * ef[j]);
            A1.v[j] = (half_t)(fd * efa[j]);
        }

        // ---- layer 1: feat(16x64) @ W1(64x64) ----
        v8f acc[4];
        #pragma unroll
        for (int nt = 0; nt < 4; ++nt) {
            v8f z = {};
            acc[nt] = wmma_f16(A0.v, B1t[0][nt], z);
            acc[nt] = wmma_f16(A1.v, B1t[1][nt], acc[nt]);
        }

        __syncthreads();   // WAR vs previous iteration's LDS reads
        #pragma unroll
        for (int nt = 0; nt < 4; ++nt)
            #pragma unroll
            for (int r = 0; r < 8; ++r) {
                float x = gelu_exact(acc[nt][r] + b1v[nt]);
                xbuf[(8 * hh + r) * HID + nt * 16 + ln] = (half_t)x;
            }
        __syncthreads();

        V16 A2[2];
        #pragma unroll
        for (int kt = 0; kt < 2; ++kt) {
            int base = ln * HID + kt * 32 + 8 * hh;
            A2[kt].h[0] = *(const v8h*)&xbuf[base];
            A2[kt].h[1] = *(const v8h*)&xbuf[base + 16];
        }

        // ---- layer 2: x(16x64) @ W2(64x64) ----
        #pragma unroll
        for (int nt = 0; nt < 4; ++nt) {
            v8f z = {};
            acc[nt] = wmma_f16(A2[0].v, B2t[0][nt], z);
            acc[nt] = wmma_f16(A2[1].v, B2t[1][nt], acc[nt]);
        }

        __syncthreads();   // WAR vs A2 loads
        #pragma unroll
        for (int nt = 0; nt < 4; ++nt)
            #pragma unroll
            for (int r = 0; r < 8; ++r) {
                float x = gelu_exact(acc[nt][r] + b2v[nt]);
                xbuf[(8 * hh + r) * HID + nt * 16 + ln] = (half_t)x;
            }
        __syncthreads();

        V16 A3[2];
        #pragma unroll
        for (int kt = 0; kt < 2; ++kt) {
            int base = ln * HID + kt * 32 + 8 * hh;
            A3[kt].h[0] = *(const v8h*)&xbuf[base];
            A3[kt].h[1] = *(const v8h*)&xbuf[base + 16];
        }

        // ---- layer 3: x(16x64) @ W3(64x32) ----
        v8f acc3[2];
        #pragma unroll
        for (int nt = 0; nt < 2; ++nt) {
            v8f z = {};
            acc3[nt] = wmma_f16(A3[0].v, B3t[0][nt], z);
            acc3[nt] = wmma_f16(A3[1].v, B3t[1][nt], acc3[nt]);
        }

        // ---- tanh * 0.25 * dec, reduce over k (lanes) ----
        // lane holds rows m = 8*hh + r for its two columns (n=ln, n=16+ln)
        float p[8];
        #pragma unroll
        for (int r = 0; r < 8; ++r) {
            p[r] = tanh_hw(acc3[0][r] + b3v[0]) * (0.25f * decv0)
                 + tanh_hw(acc3[1][r] + b3v[1]) * (0.25f * decv1);
        }
        #pragma unroll
        for (int off = 1; off <= 8; off <<= 1) {
            #pragma unroll
            for (int r = 0; r < 8; ++r)
                p[r] += __shfl_xor(p[r], off, 32);
        }
        // lanes 0 and 16 now hold fluxdot for m=0..7 and m=8..15
        if (ln == 0) {
            int mbase = i0 + 8 * hh;
            #pragma unroll
            for (int r = 0; r < 8; ++r) {
                int m = mbase + r;
                if (m < NI) fluxdot[(size_t)b * NI + m] = p[r];
            }
        }
    }
}

__global__ __launch_bounds__(256) void flux_finalize(
    const float* __restrict__ u, const float* __restrict__ enc,
    const float* __restrict__ fluxdot, float* __restrict__ out)
{
    __shared__ float Ssh;
    if (threadIdx.x == 0) {
        float s2 = 0.f;
        #pragma unroll
        for (int k = 0; k < LD; ++k) { float e = enc[k]; s2 += e * e; }
        Ssh = s2 / (s2 + 1e-12f);
    }
    __syncthreads();
    float S = Ssh;

    size_t idx = (size_t)blockIdx.x * blockDim.x + threadIdx.x;
    if (idx >= (size_t)B_SZ * N_SZ) return;
    int b = (int)(idx / N_SZ);
    int n = (int)(idx - (size_t)b * N_SZ);
    const float* fdb = fluxdot + (size_t)b * NI;
    // DT/DX = 0.0005/0.001 = 0.5
    out[idx] = u[idx] * S - 0.5f * (fdb[n + 1] - fdb[n]);
}

extern "C" void kernel_launch(void* const* d_in, const int* in_sizes, int n_in,
                              void* d_out, int out_size, void* d_ws, size_t ws_size,
                              hipStream_t stream) {
    const float* u   = (const float*)d_in[0];
    const float* enc = (const float*)d_in[1];
    const float* W1  = (const float*)d_in[2];
    const float* b1  = (const float*)d_in[3];
    const float* W2  = (const float*)d_in[4];
    const float* b2  = (const float*)d_in[5];
    const float* W3  = (const float*)d_in[6];
    const float* b3  = (const float*)d_in[7];

    float* fluxdot = (float*)d_ws;        // B*NI floats = 8.39 MB (L2-resident)
    float* out     = (float*)d_out;

    flux_mlp_pass1<<<8192, 32, 0, stream>>>(u, enc, W1, b1, W2, b2, W3, b3, fluxdot);

    int total = B_SZ * N_SZ;
    flux_finalize<<<(total + 255) / 256, 256, 0, stream>>>(u, enc, fluxdot, out);
}
